// MoEGate_35175782154751
// MI455X (gfx1250) — compile-verified, matrix operand order checked by reading, not run on hardware
//
#include <hip/hip_runtime.h>
#include <stdint.h>

// ---------------- types ----------------
typedef __attribute__((ext_vector_type(16))) __bf16 v16bf;
typedef __attribute__((ext_vector_type(8)))  __bf16 v8bf;
typedef __attribute__((ext_vector_type(8)))  float  v8f;

constexpr int kH        = 2048;   // hidden dim
constexpr int kE        = 64;     // experts
constexpr int kTopK     = 8;
constexpr int kTokPerWave = 16;
constexpr int kWaves    = 8;      // waves per block (256 threads, wave32)
constexpr int kTokPerBlock = kTokPerWave * kWaves; // 128

// ---------------- weight split: fp32 -> bf16 hi + bf16 lo ----------------
__global__ void moegate_wsplit(const float* __restrict__ w,
                               __bf16* __restrict__ whi,
                               __bf16* __restrict__ wlo, int n) {
  int i = blockIdx.x * blockDim.x + threadIdx.x;
  if (i < n) {
    float f = w[i];
    __bf16 h = (__bf16)f;              // RNE
    whi[i] = h;
    wlo[i] = (__bf16)(f - (float)h);   // residual
  }
}

// ---------------- fused gate: split-bf16 WMMA GEMM + softmax + top-8 ----------------
__launch_bounds__(256)
__global__ void moegate_main(const float*  __restrict__ x,     // [T, H] fp32, streamed NT
                             const __bf16* __restrict__ whi,   // [E, H] L2-resident
                             const __bf16* __restrict__ wlo,   // [E, H] L2-resident
                             const float*  __restrict__ bias,  // [E]
                             int*   __restrict__ out_idx,      // [T, 8]
                             float* __restrict__ out_w) {      // [T, 8]
  __shared__ float lds[kWaves][kTokPerWave][kE + 1];  // +1 pad: avoid bank conflicts
  __shared__ float lbias[kE];

  const int tid  = threadIdx.x;
  const int lane = tid & 31;
  const int wave = tid >> 5;
  if (tid < kE) lbias[tid] = bias[tid];

  const int t0   = blockIdx.x * kTokPerBlock + wave * kTokPerWave;
  const int lrow = lane & 15;     // token row (A) / expert-in-tile (B) / N column (C)
  const int sel  = lane >> 4;     // half-wave select: K offset +8 per ISA layout

  const float* xrow = x + (size_t)(t0 + lrow) * kH + sel * 8;

  v8f acc[4] = {};                // 4 N-tiles of 16 experts -> 16x64 logits

  // A double-buffer: preload K-step 0 (non-temporal: don't thrash L2 with the
  // 256MB activation stream; keep the 512KB split weights resident instead).
  v8f a0 = __builtin_nontemporal_load((const v8f*)(xrow));
  v8f a1 = __builtin_nontemporal_load((const v8f*)(xrow + 16));

  for (int k0 = 0; k0 < kH; k0 += 32) {
    // ---- B operands for ALL 4 expert tiles into distinct regs (16x b128,
    // one deep clause -> high MLP, single threshold wait before the WMMAs)
    v8bf bh0[4], bh1[4], bl0[4], bl1[4];
#pragma unroll
    for (int n = 0; n < 4; ++n) {
      const size_t ebase = (size_t)(n * 16 + lrow) * kH + k0 + sel * 8;
      bh0[n] = *(const v8bf*)(whi + ebase);
      bh1[n] = *(const v8bf*)(whi + ebase + 16);
      bl0[n] = *(const v8bf*)(wlo + ebase);
      bl1[n] = *(const v8bf*)(wlo + ebase + 16);
    }

    // ---- prefetch + preload next A chunk (branchless; last iter reloads k=0,
    // harmless L2 hit) so the HBM stream overlaps cvt + WMMA work
    const int kn = (k0 + 32 < kH) ? (k0 + 32) : 0;
    __builtin_prefetch(xrow + k0 + 64, 0, 0);   // ~2 K-steps ahead
    v8f na0 = __builtin_nontemporal_load((const v8f*)(xrow + kn));
    v8f na1 = __builtin_nontemporal_load((const v8f*)(xrow + kn + 16));

    // ---- split current A into bf16 hi/lo (ISA 16-bit A layout:
    // lanes 0-15: K {k0..k0+7, k0+16..k0+23}; lanes 16-31: +8 halves)
    v16bf ahi, alo;
#pragma unroll
    for (int j = 0; j < 8; ++j) {
      float f0 = a0[j]; __bf16 h0 = (__bf16)f0;
      ahi[j]     = h0;  alo[j]     = (__bf16)(f0 - (float)h0);
      float f1 = a1[j]; __bf16 h1 = (__bf16)f1;
      ahi[j + 8] = h1;  alo[j + 8] = (__bf16)(f1 - (float)h1);
    }

    // ---- 12 WMMAs (3 split-precision products x 4 expert tiles)
#pragma unroll
    for (int n = 0; n < 4; ++n) {
      v16bf Bhi = __builtin_shufflevector(bh0[n], bh1[n],
                    0,1,2,3,4,5,6,7,8,9,10,11,12,13,14,15);
      v16bf Blo = __builtin_shufflevector(bl0[n], bl1[n],
                    0,1,2,3,4,5,6,7,8,9,10,11,12,13,14,15);
      acc[n] = __builtin_amdgcn_wmma_f32_16x16x32_bf16(false, ahi, false, Bhi,
                                                       (short)0, acc[n], false, false);
      acc[n] = __builtin_amdgcn_wmma_f32_16x16x32_bf16(false, alo, false, Bhi,
                                                       (short)0, acc[n], false, false);
      acc[n] = __builtin_amdgcn_wmma_f32_16x16x32_bf16(false, ahi, false, Blo,
                                                       (short)0, acc[n], false, false);
    }

    a0 = na0; a1 = na1;
  }

  // ---- spill 16x64 logits to LDS per C-layout:
  // VGPR j: lanes 0-15 -> (M=j, N=lane); lanes 16-31 -> (M=j+8, N=lane-16)
#pragma unroll
  for (int n = 0; n < 4; ++n)
#pragma unroll
    for (int j = 0; j < 8; ++j)
      lds[wave][j + sel * 8][n * 16 + lrow] = acc[n][j];

  __syncthreads();

  // ---- per-token softmax + bias-adjusted top-8 (lanes 0-15 of each wave)
  if (lane < 16) {
    float* row = &lds[wave][lrow][0];
    const int token = t0 + lrow;

    float mx = -3.4e38f;
    for (int e = 0; e < kE; ++e) mx = fmaxf(mx, row[e]);
    float sum = 0.f;
    for (int e = 0; e < kE; ++e) { float v = __expf(row[e] - mx); row[e] = v; sum += v; }
    const float inv = 1.f / sum;   // scores[e] = row[e] * inv

    unsigned long long chosen = 0ull;
    float wsum = 0.f;
    int   sidx[kTopK];
    float swt[kTopK];
#pragma unroll
    for (int k = 0; k < kTopK; ++k) {
      float best = -3.4e38f; int bi = 0;
      for (int e = 0; e < kE; ++e) {
        if ((chosen >> e) & 1ull) continue;
        float s = row[e] * inv + lbias[e];       // bias-adjusted routing score
        if (s > best) { best = s; bi = e; }
      }
      chosen |= (1ull << bi);
      float wt = row[bi] * inv;                  // gather ORIGINAL softmax prob
      sidx[k] = bi; swt[k] = wt; wsum += wt;
    }
    const float r = 1.f / (wsum + 1e-20f);       // norm_topk_prob
#pragma unroll
    for (int k = 0; k < kTopK; ++k) {
      __builtin_nontemporal_store(sidx[k],     &out_idx[(size_t)token * kTopK + k]);
      __builtin_nontemporal_store(swt[k] * r,  &out_w [(size_t)token * kTopK + k]);
    }
  }
}

// ---------------- launcher ----------------
extern "C" void kernel_launch(void* const* d_in, const int* in_sizes, int n_in,
                              void* d_out, int out_size, void* d_ws, size_t ws_size,
                              hipStream_t stream) {
  const float* x    = (const float*)d_in[0];   // [8,4096,2048]
  const float* w    = (const float*)d_in[1];   // [64,2048]
  const float* bias = (const float*)d_in[2];   // [64]
  // d_in[3] = top_k (==8, baked in)

  const int T = in_sizes[0] / kH;              // 32768 tokens

  __bf16* whi = (__bf16*)d_ws;
  __bf16* wlo = whi + (size_t)kE * kH;         // 512 KB total in workspace

  int*   out_idx = (int*)d_out;                       // first T*8: int32 indices
  float* out_w   = (float*)d_out + (size_t)T * kTopK; // then T*8: fp32 weights

  const int nw = kE * kH;
  moegate_wsplit<<<(nw + 255) / 256, 256, 0, stream>>>(w, whi, wlo, nw);
  moegate_main<<<T / kTokPerBlock, 256, 0, stream>>>(x, whi, wlo, bias, out_idx, out_w);
}
